// ST_51531017617487
// MI455X (gfx1250) — matrix operand order repaired and checked by reading, not compiled, not run on hardware
//
#include <hip/hip_runtime.h>
#include <hip/hip_bf16.h>
#include <math.h>

// ---------------------------------------------------------------------------
// VGAE forward for MI455X (gfx1250, wave32, WMMA).
// Dominant cost: readj = sigmoid(emb @ emb.T) -> 10000x10000 f32 = 400 MB
// stores => HBM bound (~19us floor @ 23.3 TB/s). All dense matmuls use
// v_wmma_f32_16x16x32_f16 (f16 operands, f32 accumulate). emb is converted
// to f16 once (1.25 MB, L2-resident) so the readj inner loop moves only
// 16-bit operands: A fragments = 2x b128 global loads/lane, B tile staged
// to LDS via async-load-to-LDS (ASYNCcnt). GEMM operand staging is done in
// float4/b128 chunks so loads batch and overlap (no per-element waits).
// ---------------------------------------------------------------------------

#define N_NODES  10000
#define E_EDGES  320000
#define NFEAT    512
#define NHID1    256
#define NHID2    64
#define H1       128      // 2*NHID2
#define NEG_SLOPE 0.2f
#define BN_EPS   1e-5f
#define BRS      40       // padded LDS row stride in halves: 80 B, 16B-aligned,
                          // bank-conflict-free (n*20 mod 64 distinct, n<16)

typedef __attribute__((ext_vector_type(16))) _Float16 v16h;
typedef __attribute__((ext_vector_type(8)))  _Float16 v8h;
typedef __attribute__((ext_vector_type(4)))  _Float16 v4h;
typedef __attribute__((ext_vector_type(8)))  float    v8f;
typedef __attribute__((ext_vector_type(4)))  float    v4f;
typedef __attribute__((ext_vector_type(2)))  int      v2i;

// Address-space qualified pointer types for the async-to-LDS builtin:
// clang's prototype wants (AS1 v2i*, AS3 v2i*, imm offset, imm cpol).
typedef __attribute__((address_space(1))) v2i* gptr_v2i;
typedef __attribute__((address_space(3))) v2i* lptr_v2i;

#if defined(__has_builtin)
#if __has_builtin(__builtin_amdgcn_global_load_async_to_lds_b64)
#define HAVE_ASYNC_LDS 1
#endif
#if __has_builtin(__builtin_amdgcn_s_wait_asynccnt)
#define HAVE_WAIT_ASYNC 1
#endif
#endif

__device__ inline void wait_async0() {
#if defined(HAVE_ASYNC_LDS)
#if defined(HAVE_WAIT_ASYNC)
    __builtin_amdgcn_s_wait_asynccnt(0);
#else
    asm volatile("s_wait_asynccnt 0" ::: "memory");
#endif
#endif
}

// ------------------------------ small helpers ------------------------------

__global__ void fill_val_kernel(float* __restrict__ p, float v, int n) {
    int i = blockIdx.x * blockDim.x + threadIdx.x;
    if (i < n) p[i] = v;
}

__global__ void fill_bias_rows_kernel(float* __restrict__ p,
                                      const float* __restrict__ bias,
                                      int rows, int cols) {
    int i = blockIdx.x * blockDim.x + threadIdx.x;
    if (i < rows * cols) p[i] = bias[i % cols];
}

__device__ inline void atomicMaxF(float* addr, float val) {
    // IEEE monotonic-punning float atomic max (init must be -inf).
    if (val >= 0.0f) atomicMax((int*)addr, __float_as_int(val));
    else             atomicMin((unsigned int*)addr, __float_as_uint(val));
}

// ------------------------- generic WMMA GEMM + bias ------------------------
// C[M,Ncol] = A[M,K] @ W[K,Ncol] + bias. Requirements (all met here):
//   M % 16 == 0, K % 32 == 0, Ncol % 64 == 0.
// Block = 128 threads (4 waves), tile = 16 rows x 64 cols. Both operands are
// staged through LDS as f16 in b128 global chunks: A tile 16x32 (k-major),
// W tile 32x64 (n-major, padded stride) so every lane's A/B fragment is
// contiguous b128 ds loads.
__global__ void __launch_bounds__(128)
gemm_bias_wmma_kernel(const float* __restrict__ A, const float* __restrict__ W,
                      const float* __restrict__ bias, float* __restrict__ Cout,
                      int M, int K, int Ncol) {
    __shared__ _Float16 sA[16 * 32];
    __shared__ _Float16 sB[64 * BRS];

    const int lane = threadIdx.x & 31;
    const int wave = threadIdx.x >> 5;
    const int m0   = blockIdx.x * 16;
    const int nblk = blockIdx.y * 64;

    const int mrow = lane & 15;            // A fragment row
    const int kbA  = (lane >> 4) * 8;      // A fragment K base (ISA f16-A layout)
    const int nn   = wave * 16 + (lane & 15);
    const int ncol = nblk + nn;            // B/C fragment column
    const int kbB  = (lane >> 4) * 16;     // unused: B fragment uses kbA (same K order as A)

    v8f acc = {};
    for (int k0 = 0; k0 < K; k0 += 32) {
        __syncthreads();
        // Stage A tile [16 x 32]: one float4 (b128) per thread, packed f16
        // convert, single 8-byte LDS store. (No row crossing: 32 % 4 == 0.)
        {
            int e  = threadIdx.x * 4;                      // 0..508
            int r  = e >> 5;
            int kk = e & 31;
            v4f a4 = *(const v4f*)&A[(m0 + r) * K + k0 + kk];
            v4h h4;
            #pragma unroll
            for (int i = 0; i < 4; ++i) h4[i] = (_Float16)a4[i];
            *(v4h*)&sA[e] = h4;
        }
        // Stage W tile [32 x 64] n-major/padded: 4x float4 (b128) per thread,
        // all loads issued before use (batched, overlapping).
        #pragma unroll
        for (int j = 0; j < 4; ++j) {
            int e  = j * 512 + threadIdx.x * 4;            // 0..2044
            int kk = e >> 6;
            int cc = e & 63;
            v4f w4 = *(const v4f*)&W[(k0 + kk) * Ncol + nblk + cc];
            #pragma unroll
            for (int i = 0; i < 4; ++i) sB[(cc + i) * BRS + kk] = (_Float16)w4[i];
        }
        __syncthreads();

        if (k0 + 32 < K)   // prefetch next A stripe (global_prefetch_b8)
            __builtin_prefetch(&A[(m0 + (threadIdx.x & 15)) * K + k0 + 32], 0, 1);

        // A fragment: two contiguous 8-half runs per lane (16B aligned).
        v16h af, bf;
        const _Float16* ap = &sA[mrow * 32];
        #pragma unroll
        for (int i = 0; i < 8; ++i) { af[i] = ap[kbA + i]; af[i + 8] = ap[16 + kbA + i]; }
        // B fragment: two contiguous 8-half runs per lane, same K order as A (16B aligned).
        const _Float16* bp = &sB[nn * BRS + kbA];
        #pragma unroll
        for (int i = 0; i < 8; ++i) { bf[i] = bp[i]; bf[i + 8] = bp[16 + i]; }

        acc = __builtin_amdgcn_wmma_f32_16x16x32_f16(
            false, af, false, bf, (short)0, acc, false, false);
    }

    const float bv = bias ? bias[ncol] : 0.0f;
    #pragma unroll
    for (int r = 0; r < 8; ++r) {
        int row = m0 + r + 8 * (lane >> 4);   // C layout: VGPR r -> M=r (+8 hi lanes)
        Cout[(size_t)row * Ncol + ncol] = acc[r] + bv;
    }
}

// --------------------- readj = sigmoid(emb @ emb^T) ------------------------
// emb pre-converted to f16. Block = 4 waves stacked in M (64x16 tile). The
// shared 32x16-half B tile is a raw byte copy of 16 emb_h row segments ->
// async global->LDS b64 copies (ASYNCcnt). A fragments: 2x b128 global loads
// of halves per lane. Row edge handled by clamped loads (EXEC all-1s for
// WMMA) + predicated stores.
__global__ void __launch_bounds__(128)
readj_wmma_kernel(const _Float16* __restrict__ embh, float* __restrict__ out, int Nn) {
    __shared__ _Float16 sB[16 * BRS];

    const int lane = threadIdx.x & 31;
    const int wave = threadIdx.x >> 5;
    const int m0   = blockIdx.x * 64 + wave * 16;
    const int n0   = blockIdx.y * 16;

    const int mrow = lane & 15;
    const int kbA  = (lane >> 4) * 8;
    const int kbB  = (lane >> 4) * 16;

    int arow = m0 + mrow;
    if (arow >= Nn) arow = Nn - 1;            // clamp: keep EXEC all-1s

    // Staging assignment: thread -> (row, 8-byte segment of 64B row chunk).
    const int srow = threadIdx.x >> 3;        // 0..15
    const int sseg = threadIdx.x & 7;         // 0..7

    v8f acc = {};
    for (int k0 = 0; k0 < NHID2; k0 += 32) {
        __syncthreads();
        {
            const _Float16* g = embh + (n0 + srow) * NHID2 + k0 + sseg * 4;
            _Float16*       l = &sB[srow * BRS + sseg * 4];
#if defined(HAVE_ASYNC_LDS)
            __builtin_amdgcn_global_load_async_to_lds_b64(
                (gptr_v2i)g, (lptr_v2i)l, 0, 0);
            wait_async0();
#else
            #pragma unroll
            for (int j = 0; j < 4; ++j) l[j] = g[j];
#endif
        }
        __syncthreads();

        // A fragment: two aligned v8h loads from the f16 emb row.
        v16h af, bf;
        const v8h* rp = (const v8h*)(embh + arow * NHID2 + k0);
        v8h lo = rp[kbA >> 3];
        v8h hi = rp[2 + (kbA >> 3)];
        #pragma unroll
        for (int i = 0; i < 8; ++i) { af[i] = lo[i]; af[i + 8] = hi[i]; }
        // B fragment: two 8-half runs per lane from LDS, same K order as A (b128 loads).
        const _Float16* bp = &sB[(lane & 15) * BRS + kbA];
        #pragma unroll
        for (int i = 0; i < 8; ++i) { bf[i] = bp[i]; bf[i + 8] = bp[16 + i]; }

        acc = __builtin_amdgcn_wmma_f32_16x16x32_f16(
            false, af, false, bf, (short)0, acc, false, false);
    }

    const int ncol = n0 + (lane & 15);
    #pragma unroll
    for (int r = 0; r < 8; ++r) {
        int row = m0 + r + 8 * (lane >> 4);
        if (row < Nn)
            out[(size_t)row * Nn + ncol] = 1.0f / (1.0f + __expf(-acc[r]));
    }
}

// ----------------------------- GATv2 edge passes ---------------------------

__global__ void edge_logits_kernel(const float* __restrict__ xl,
                                   const float* __restrict__ xr,
                                   const float* __restrict__ att,
                                   const int* __restrict__ ei,
                                   int E, int Nn, int C,
                                   float* __restrict__ ee,
                                   float* __restrict__ emax) {
    int i = blockIdx.x * blockDim.x + threadIdx.x;
    int tot = E + Nn;
    if (i >= tot) return;
    int s, t;
    if (i < E) { s = ei[i]; t = ei[E + i]; } else { s = t = i - E; }
    float acc = 0.0f;
    for (int c = 0; c < C; ++c) {
        float v = xl[s * C + c] + xr[t * C + c];
        v = (v > 0.0f) ? v : NEG_SLOPE * v;     // leaky_relu
        acc += att[c] * v;
    }
    ee[i] = acc;
    atomicMaxF(&emax[t], acc);
}

__global__ void edge_expsum_kernel(const float* __restrict__ ee,
                                   const float* __restrict__ emax,
                                   const int* __restrict__ ei,
                                   int E, int Nn,
                                   float* __restrict__ ex,
                                   float* __restrict__ esum) {
    int i = blockIdx.x * blockDim.x + threadIdx.x;
    int tot = E + Nn;
    if (i >= tot) return;
    int t = (i < E) ? ei[E + i] : (i - E);
    float v = __expf(ee[i] - emax[t]);
    ex[i] = v;
    atomicAdd(&esum[t], v);
}

__global__ void edge_aggregate_kernel(const float* __restrict__ ex,
                                      const float* __restrict__ esum,
                                      const float* __restrict__ xl,
                                      const int* __restrict__ ei,
                                      int E, int Nn, int C,
                                      float* __restrict__ out) {
    int id = blockIdx.x * blockDim.x + threadIdx.x;   // (E+Nn)*C <= 42.2M, fits int
    int tot = (E + Nn) * C;
    if (id >= tot) return;
    int i = id / C;
    int c = id - i * C;
    int s, t;
    if (i < E) { s = ei[i]; t = ei[E + i]; } else { s = t = i - E; }
    float alpha = ex[i] / esum[t];
    atomicAdd(&out[t * C + c], alpha * xl[s * C + c]);
}

// ------------------------- batch norm (+ReLU) in place ---------------------
__global__ void __launch_bounds__(256)
bn_relu_kernel(float* __restrict__ h, const float* __restrict__ g,
               const float* __restrict__ b, int rows, int cols, int doRelu) {
    __shared__ float ssum[256];
    __shared__ float ssq[256];
    const int col = blockIdx.x;
    float s = 0.0f, q = 0.0f;
    for (int r = threadIdx.x; r < rows; r += 256) {
        float v = h[(size_t)r * cols + col];
        s += v; q += v * v;
    }
    ssum[threadIdx.x] = s; ssq[threadIdx.x] = q;
    __syncthreads();
    for (int st = 128; st > 0; st >>= 1) {
        if (threadIdx.x < st) {
            ssum[threadIdx.x] += ssum[threadIdx.x + st];
            ssq[threadIdx.x]  += ssq[threadIdx.x + st];
        }
        __syncthreads();
    }
    const float mean  = ssum[0] / (float)rows;
    const float var   = ssq[0] / (float)rows - mean * mean;
    const float scale = rsqrtf(var + BN_EPS) * g[col];
    const float shift = b[col];
    for (int r = threadIdx.x; r < rows; r += 256) {
        float v = (h[(size_t)r * cols + col] - mean) * scale + shift;
        if (doRelu) v = fmaxf(v, 0.0f);
        h[(size_t)r * cols + col] = v;
    }
}

// ---------------- reparameterize (+ f16 cast of emb for readj) -------------

__global__ void reparam_kernel(const float* __restrict__ eps_,
                               const float* __restrict__ mu,
                               const float* __restrict__ logvar,
                               float* __restrict__ emb,
                               _Float16* __restrict__ embh, int n) {
    int i = blockIdx.x * blockDim.x + threadIdx.x;
    if (i < n) {
        float v = eps_[i] * __expf(logvar[i]) + mu[i];
        emb[i]  = v;
        embh[i] = (_Float16)v;
    }
}

// ------------------------------- orchestration -----------------------------

static inline void run_gemm(const float* A, const float* W, const float* bias,
                            float* C, int M, int K, int Ncol, hipStream_t s) {
    dim3 grid(M / 16, Ncol / 64);
    gemm_bias_wmma_kernel<<<grid, 128, 0, s>>>(A, W, bias, C, M, K, Ncol);
}

static inline void run_gat_edges(const float* xl, const float* xr, const float* att,
                                 const int* ei, int C, float* out_acc,
                                 float* ee, float* ex, float* emax, float* esum,
                                 hipStream_t s) {
    const int tot = E_EDGES + N_NODES;
    fill_val_kernel<<<(N_NODES + 255) / 256, 256, 0, s>>>(emax, -INFINITY, N_NODES);
    fill_val_kernel<<<(N_NODES + 255) / 256, 256, 0, s>>>(esum, 0.0f, N_NODES);
    edge_logits_kernel<<<(tot + 255) / 256, 256, 0, s>>>(xl, xr, att, ei,
                                                         E_EDGES, N_NODES, C, ee, emax);
    edge_expsum_kernel<<<(tot + 255) / 256, 256, 0, s>>>(ee, emax, ei,
                                                         E_EDGES, N_NODES, ex, esum);
    edge_aggregate_kernel<<<(tot * C + 255) / 256, 256, 0, s>>>(ex, esum, xl, ei,
                                                                E_EDGES, N_NODES, C, out_acc);
}

extern "C" void kernel_launch(void* const* d_in, const int* in_sizes, int n_in,
                              void* d_out, int out_size, void* d_ws, size_t ws_size,
                              hipStream_t stream) {
    // ---- inputs (setup_inputs dict order) ----
    const float* x      = (const float*)d_in[0];
    const int*   ei     = (const int*)  d_in[1];
    const float* eps_   = (const float*)d_in[2];
    const float* w1l    = (const float*)d_in[3];
    const float* b1l    = (const float*)d_in[4];
    const float* w1r    = (const float*)d_in[5];
    const float* b1r    = (const float*)d_in[6];
    const float* a1     = (const float*)d_in[7];
    const float* bias1  = (const float*)d_in[8];
    const float* bn1g   = (const float*)d_in[9];
    const float* bn1b   = (const float*)d_in[10];
    const float* w2l    = (const float*)d_in[11];
    const float* b2l    = (const float*)d_in[12];
    const float* w2r    = (const float*)d_in[13];
    const float* b2r    = (const float*)d_in[14];
    const float* a2     = (const float*)d_in[15];
    const float* bias2  = (const float*)d_in[16];
    const float* w3l    = (const float*)d_in[17];
    const float* b3l    = (const float*)d_in[18];
    const float* w3r    = (const float*)d_in[19];
    const float* b3r    = (const float*)d_in[20];
    const float* a3     = (const float*)d_in[21];
    const float* bias3  = (const float*)d_in[22];
    const float* dw1    = (const float*)d_in[23];
    const float* db1    = (const float*)d_in[24];
    const float* dbng   = (const float*)d_in[25];
    const float* dbnb   = (const float*)d_in[26];
    const float* dw2    = (const float*)d_in[27];
    const float* db2    = (const float*)d_in[28];
    (void)in_sizes; (void)n_in; (void)out_size; (void)ws_size;

    // ---- output slices (return order: emb, re_x, readj, mu, logvar) ----
    float* out    = (float*)d_out;
    float* emb    = out;                                   // N*64
    float* re_x   = out + (size_t)N_NODES * NHID2;         // N*512
    float* readj  = re_x + (size_t)N_NODES * NFEAT;        // N*N
    float* mu     = readj + (size_t)N_NODES * N_NODES;     // N*64
    float* logvar = mu + (size_t)N_NODES * NHID2;          // N*64

    // ---- workspace layout ----
    float* ws   = (float*)d_ws;
    float* xl1  = ws;                              // N*H1
    float* xr1  = xl1 + (size_t)N_NODES * H1;      // N*H1
    float* h    = xr1 + (size_t)N_NODES * H1;      // N*H1
    float* xl2  = h   + (size_t)N_NODES * H1;      // N*64
    float* xr2  = xl2 + (size_t)N_NODES * NHID2;
    float* xl3  = xr2 + (size_t)N_NODES * NHID2;
    float* xr3  = xl3 + (size_t)N_NODES * NHID2;
    float* dhid = xr3 + (size_t)N_NODES * NHID2;   // N*NHID1
    float* ee   = dhid + (size_t)N_NODES * NHID1;  // E+N
    float* ex   = ee + (E_EDGES + N_NODES);        // E+N
    float* emax = ex + (E_EDGES + N_NODES);        // N
    float* esum = emax + N_NODES;                  // N
    _Float16* embh = (_Float16*)(esum + N_NODES);  // N*64 halves

    // ---- GATv2 layer 1: nfeat(512) -> 128 ----
    run_gemm(x, w1l, b1l, xl1, N_NODES, NFEAT, H1, stream);
    run_gemm(x, w1r, b1r, xr1, N_NODES, NFEAT, H1, stream);
    fill_bias_rows_kernel<<<(N_NODES * H1 + 255) / 256, 256, 0, stream>>>(h, bias1, N_NODES, H1);
    run_gat_edges(xl1, xr1, a1, ei, H1, h, ee, ex, emax, esum, stream);
    bn_relu_kernel<<<H1, 256, 0, stream>>>(h, bn1g, bn1b, N_NODES, H1, 1);

    // ---- GATv2 layer 2 (mu) and 3 (logvar): 128 -> 64 ----
    run_gemm(h, w2l, b2l, xl2, N_NODES, H1, NHID2, stream);
    run_gemm(h, w2r, b2r, xr2, N_NODES, H1, NHID2, stream);
    run_gemm(h, w3l, b3l, xl3, N_NODES, H1, NHID2, stream);
    run_gemm(h, w3r, b3r, xr3, N_NODES, H1, NHID2, stream);

    fill_bias_rows_kernel<<<(N_NODES * NHID2 + 255) / 256, 256, 0, stream>>>(mu, bias2, N_NODES, NHID2);
    run_gat_edges(xl2, xr2, a2, ei, NHID2, mu, ee, ex, emax, esum, stream);

    fill_bias_rows_kernel<<<(N_NODES * NHID2 + 255) / 256, 256, 0, stream>>>(logvar, bias3, N_NODES, NHID2);
    run_gat_edges(xl3, xr3, a3, ei, NHID2, logvar, ee, ex, emax, esum, stream);

    // ---- reparameterize: emb = eps * exp(logvar) + mu (+ f16 copy) ----
    reparam_kernel<<<(N_NODES * NHID2 + 255) / 256, 256, 0, stream>>>(
        eps_, mu, logvar, emb, embh, N_NODES * NHID2);

    // ---- readj = sigmoid(emb @ emb^T): the HBM-bound 400 MB store ----
    {
        dim3 grid((N_NODES + 63) / 64, N_NODES / 16);
        readj_wmma_kernel<<<grid, 128, 0, stream>>>(embh, readj, N_NODES);
    }

    // ---- decoder MLP: 64 -> 256 (BN+ReLU) -> 512 ----
    run_gemm(emb, dw1, db1, dhid, N_NODES, NHID2, NHID1, stream);
    bn_relu_kernel<<<NHID1, 256, 0, stream>>>(dhid, dbng, dbnb, N_NODES, NHID1, 1);
    run_gemm(dhid, dw2, db2, re_x, N_NODES, NHID1, NFEAT, stream);
}
